// MultiDilateLocalAttention_44624710205603
// MI455X (gfx1250) — compile-verified, matrix operand order checked by reading, not run on hardware
//
#include <hip/hip_runtime.h>

typedef __attribute__((ext_vector_type(8)))  float  v8f;
typedef __attribute__((ext_vector_type(8)))  __bf16 v8bf;
typedef __attribute__((ext_vector_type(16))) __bf16 v16bf;

typedef unsigned int       u32;
typedef unsigned short     u16;
typedef unsigned long long u64;

__device__ __forceinline__ u16 f2bf(float f) {
    u32 u = __float_as_uint(f);
    u += 0x7FFFu + ((u >> 16) & 1u);      // round-to-nearest-even
    return (u16)(u >> 16);
}
__device__ __forceinline__ float bf2f(u16 h) {
    return __uint_as_float(((u32)h) << 16);
}

// LDS generic pointer -> wave-relative LDS byte offset (low 32 bits of the
// shared-aperture generic address are the LDS offset on gfx1250).
__device__ __forceinline__ u32 lds_off(const void* p) {
    return (u32)(u64)p;
}

// async global -> LDS copy of 16 bytes per lane; INST_OFFSET applies to both
// the LDS and global addresses, so one base covers consecutive chunks.
#define ASYNC_CP_B128(ldsaddr, gaddr, imm)                                    \
    asm volatile("global_load_async_to_lds_b128 %0, %1, off offset:" #imm     \
                 :: "v"(ldsaddr), "v"(gaddr) : "memory")
#define WAIT_ASYNC() asm volatile("s_wait_asynccnt 0" ::: "memory")

// ------------------------------------------------------------------
// elementwise f32 -> bf16 cast (4 elems / thread)
// ------------------------------------------------------------------
__global__ __launch_bounds__(256)
void cast_f32_to_bf16(const float* __restrict__ in, u16* __restrict__ out, int n4) {
    int i = blockIdx.x * blockDim.x + threadIdx.x;
    if (i >= n4) return;
    float4 v = ((const float4*)in)[i];
    u32 lo = (u32)f2bf(v.x) | ((u32)f2bf(v.y) << 16);
    u32 hi = (u32)f2bf(v.z) | ((u32)f2bf(v.w) << 16);
    ((uint2*)out)[i] = make_uint2(lo, hi);
}

// ------------------------------------------------------------------
// weight transpose + cast: in is [K][N] row-major f32, out is [N][K] bf16
// ------------------------------------------------------------------
__global__ void transpose_cast(const float* __restrict__ in, u16* __restrict__ out,
                               int K, int N) {
    int n = blockIdx.x * 16 + threadIdx.x;
    int k = blockIdx.y * 16 + threadIdx.y;
    if (n < N && k < K)
        out[(size_t)n * K + k] = f2bf(in[(size_t)k * N + n]);
}

// ------------------------------------------------------------------
// WMMA bf16 GEMM:  C[M][N] = A[M][K] * B^T   (B stored [N][K] bf16)
//   block tile 128x256, BK=32, 8 waves (2 x 4), wave tile 64x64
//   double-buffered LDS, next tile staged with GLOBAL_LOAD_ASYNC_TO_LDS_B128
//   AF32:  A supplied as f32 (converted to bf16 while staging to LDS)
//   OUTF32: write f32 + bias, else write bf16
// ------------------------------------------------------------------
#define BM 128
#define BN 256
#define BK 32
#define LDT 56   // LDS row stride (bf16): 112 B, 16B-aligned, conflict-free 28-bank stride

template<bool AF32, bool OUTF32>
__global__ __launch_bounds__(256)
void wmma_gemm(const u16* __restrict__ Ab, const float* __restrict__ Af,
               const u16* __restrict__ Bt,
               u16* __restrict__ Cb, float* __restrict__ Cf,
               const float* __restrict__ bias,
               int K, int N)
{
    __shared__ u16 As[2][BM * LDT];   // 2 x 14,336 B
    __shared__ u16 Bs[2][BN * LDT];   // 2 x 28,672 B

    const int tid  = threadIdx.x;
    const int lane = tid & 31;
    const int wave = tid >> 5;
    const int wm   = wave & 1;      // 0..1 : 64-row slab
    const int wn   = wave >> 1;     // 0..3 : 64-col slab
    const int half = lane >> 4;
    const int lm   = lane & 15;

    const int m0 = blockIdx.x * BM;
    const int n0 = blockIdx.y * BN;

    const int rA = tid >> 1;            // 0..127 : A staging row
    const int cA = (tid & 1) * 16;      // 0 / 16 within BK

    v8f accv[4][4];
    const v8f vzero = {0.f,0.f,0.f,0.f,0.f,0.f,0.f,0.f};
    #pragma unroll
    for (int i = 0; i < 4; ++i)
        #pragma unroll
        for (int j = 0; j < 4; ++j)
            accv[i][j] = vzero;

    // ---- stage one K-tile into LDS buffer `buf` ----
    auto stage = [&](int buf, int k0) {
        if (AF32) {
            const float* g = Af + (size_t)(m0 + rA) * K + k0 + cA;
            u32 p[8];
            #pragma unroll
            for (int t = 0; t < 8; ++t)
                p[t] = (u32)f2bf(g[2*t]) | ((u32)f2bf(g[2*t+1]) << 16);
            *(uint4*)&As[buf][rA * LDT + cA]     = make_uint4(p[0], p[1], p[2], p[3]);
            *(uint4*)&As[buf][rA * LDT + cA + 8] = make_uint4(p[4], p[5], p[6], p[7]);
            if (k0 + BK < K) __builtin_prefetch(g + BK, 0, 3);
        } else {
            const u16* g = Ab + (size_t)(m0 + rA) * K + k0 + cA;
            u32 la = lds_off(&As[buf][rA * LDT + cA]);
            ASYNC_CP_B128(la, g, 0);
            ASYNC_CP_B128(la, g, 16);
        }
        // B tile: thread t stages row t (32 bf16 = 4 x b128 chunks)
        {
            const u16* g = Bt + (size_t)(n0 + tid) * K + k0;
            u32 lb = lds_off(&Bs[buf][tid * LDT]);
            ASYNC_CP_B128(lb, g, 0);
            ASYNC_CP_B128(lb, g, 16);
            ASYNC_CP_B128(lb, g, 32);
            ASYNC_CP_B128(lb, g, 48);
        }
    };

    stage(0, 0);
    WAIT_ASYNC();
    __syncthreads();

    int buf = 0;
    for (int k0 = 0; k0 < K; k0 += BK) {
        const bool more = (k0 + BK) < K;
        if (more) stage(buf ^ 1, k0 + BK);   // async copy overlaps compute below

        // -------- fragment loads (ISA 16-bit layout: lanes 0-15 K0..7/16..23,
        //          lanes 16-31 +8; two aligned ds_load_b128 each) --------
        v16bf afr[4], bfr[4];
        #pragma unroll
        for (int i = 0; i < 4; ++i) {
            const u16* p = &As[buf][(wm * 64 + i * 16 + lm) * LDT + half * 8];
            v8bf lo = *(const v8bf*)p;
            v8bf hi = *(const v8bf*)(p + 16);
            afr[i] = __builtin_shufflevector(lo, hi,
                        0,1,2,3,4,5,6,7,8,9,10,11,12,13,14,15);
        }
        #pragma unroll
        for (int j = 0; j < 4; ++j) {
            const u16* p = &Bs[buf][(wn * 64 + j * 16 + lm) * LDT + half * 8];
            v8bf lo = *(const v8bf*)p;
            v8bf hi = *(const v8bf*)(p + 16);
            bfr[j] = __builtin_shufflevector(lo, hi,
                        0,1,2,3,4,5,6,7,8,9,10,11,12,13,14,15);
        }

        #pragma unroll
        for (int i = 0; i < 4; ++i)
            #pragma unroll
            for (int j = 0; j < 4; ++j)
                accv[i][j] = __builtin_amdgcn_wmma_f32_16x16x32_bf16(
                    false, afr[i], false, bfr[j], (short)0, accv[i][j],
                    false, false);

        if (more) WAIT_ASYNC();              // next tile's LDS writes complete
        __syncthreads();
        buf ^= 1;
    }

    // -------- epilogue: C/D layout -> VGPR t holds row (t + 8*half) --------
    #pragma unroll
    for (int i = 0; i < 4; ++i) {
        #pragma unroll
        for (int j = 0; j < 4; ++j) {
            const int gm = m0 + wm * 64 + i * 16 + half * 8;
            const int gn = n0 + wn * 64 + j * 16 + lm;
            if (OUTF32) {
                const float bv = bias[gn];
                #pragma unroll
                for (int t = 0; t < 8; ++t)
                    Cf[(size_t)(gm + t) * N + gn] = accv[i][j][t] + bv;
            } else {
                #pragma unroll
                for (int t = 0; t < 8; ++t)
                    Cb[(size_t)(gm + t) * N + gn] = f2bf(accv[i][j][t]);
            }
        }
    }
}

// ------------------------------------------------------------------
// 3-tap dilated local attention, one wave per token, 8 channels/lane.
// qkv: [M][768] bf16 (q | k | v).  Zero-pad positions keep logit == 0
// and contribute exp(0) to the softmax denominator (matches jnp.pad).
// acc += (1/3) * branch output (or = for the first dilation).
// ------------------------------------------------------------------
__global__ __launch_bounds__(256)
void attn_kernel(const u16* __restrict__ qkv, float* __restrict__ acc,
                 int dil, float wsum, int accumulate)
{
    const int lane = threadIdx.x & 31;
    const int wave = threadIdx.x >> 5;
    const int m = blockIdx.x * 8 + wave;    // token id (b*4096 + n)
    const int b = m >> 12;
    const int n = m & 4095;
    const int c0 = lane * 8;

    float q[8];
    {
        uint4 d = *(const uint4*)(qkv + (size_t)m * 768 + c0);
        u32 w[4] = {d.x, d.y, d.z, d.w};
        #pragma unroll
        for (int t = 0; t < 4; ++t) {
            q[2*t]   = bf2f((u16)(w[t] & 0xFFFFu));
            q[2*t+1] = bf2f((u16)(w[t] >> 16));
        }
    }

    float l[3];
    float vv[3][8];
    #pragma unroll
    for (int j = 0; j < 3; ++j) {
        const int nn = n + (j - 1) * dil;
        const bool ok = (unsigned)nn < 4096u;      // uniform across the wave
        float p = 0.0f;
        #pragma unroll
        for (int t = 0; t < 8; ++t) vv[j][t] = 0.0f;
        if (ok) {
            const u16* row = qkv + ((size_t)((b << 12) + nn)) * 768 + c0;
            uint4 kd = *(const uint4*)(row + 256);
            uint4 vd = *(const uint4*)(row + 512);
            u32 kw[4] = {kd.x, kd.y, kd.z, kd.w};
            u32 vw[4] = {vd.x, vd.y, vd.z, vd.w};
            #pragma unroll
            for (int t = 0; t < 4; ++t) {
                p += q[2*t]   * bf2f((u16)(kw[t] & 0xFFFFu));
                p += q[2*t+1] * bf2f((u16)(kw[t] >> 16));
                vv[j][2*t]   = bf2f((u16)(vw[t] & 0xFFFFu));
                vv[j][2*t+1] = bf2f((u16)(vw[t] >> 16));
            }
        }
        #pragma unroll
        for (int off = 16; off >= 1; off >>= 1)
            p += __shfl_xor(p, off, 32);
        l[j] = p * 0.0625f;                       // scale = 256^-0.5
    }

    const float mx = fmaxf(l[0], fmaxf(l[1], l[2]));
    const float e0 = __expf(l[0] - mx);
    const float e1 = __expf(l[1] - mx);
    const float e2 = __expf(l[2] - mx);
    const float inv = wsum / (e0 + e1 + e2);
    const float a0 = e0 * inv, a1 = e1 * inv, a2 = e2 * inv;

    float* dst = acc + (size_t)m * 256 + c0;
    #pragma unroll
    for (int t = 0; t < 8; ++t) {
        const float o = a0 * vv[0][t] + a1 * vv[1][t] + a2 * vv[2][t];
        if (accumulate) dst[t] += o;
        else            dst[t] = o;
    }
}

// ------------------------------------------------------------------
extern "C" void kernel_launch(void* const* d_in, const int* in_sizes, int n_in,
                              void* d_out, int out_size, void* d_ws, size_t ws_size,
                              hipStream_t stream)
{
    (void)in_sizes; (void)n_in; (void)out_size; (void)ws_size;
    const float* x     = (const float*)d_in[0];   // (16,4096,256)
    const float* Wqkv  = (const float*)d_in[1];   // (3,256,768)
    const float* Wproj = (const float*)d_in[2];   // (256,256)
    const float* bproj = (const float*)d_in[3];   // (256,)
    float* out = (float*)d_out;                   // (16,4096,256) f32

    const int C  = 256;
    const int M  = 16 * 4096;     // 65536 tokens
    const int NQ = 3 * C;         // 768

    // workspace layout (all offsets 256-B aligned)
    char* ws = (char*)d_ws;
    u16*   xb     = (u16*)(ws);                       //  32 MB  x as bf16
    u16*   wqkvT  = (u16*)(ws + 33554432);            //  1.125 MB [3][768][256] bf16
    u16*   wprojT = (u16*)(ws + 34734080);            //  128 KB [256][256] bf16
    u16*   qkvb   = (u16*)(ws + 34865152);            //  96 MB  [M][768] bf16
    float* accp   = (float*)(ws + 135528448);         //  64 MB  [M][256] f32

    // 1) cast x to bf16
    {
        const int n4 = M * C / 4;
        cast_f32_to_bf16<<<(n4 + 255) / 256, 256, 0, stream>>>(x, xb, n4);
    }

    // 2) transpose+cast weights to [N][K] bf16
    {
        dim3 tb(16, 16);
        for (int i = 0; i < 3; ++i)
            transpose_cast<<<dim3(NQ / 16, C / 16), tb, 0, stream>>>(
                Wqkv + (size_t)i * C * NQ, wqkvT + (size_t)i * NQ * C, C, NQ);
        transpose_cast<<<dim3(C / 16, C / 16), tb, 0, stream>>>(Wproj, wprojT, C, C);
    }

    // 3) per-dilation: QKV GEMM (WMMA) then local attention accumulate
    const int dil[3] = {1, 2, 3};
    for (int i = 0; i < 3; ++i) {
        wmma_gemm<false, false><<<dim3(M / BM, NQ / BN), 256, 0, stream>>>(
            xb, nullptr, wqkvT + (size_t)i * NQ * C,
            qkvb, nullptr, nullptr, C, NQ);
        attn_kernel<<<M / 8, 256, 0, stream>>>(qkvb, accp, dil[i], 1.0f / 3.0f, i);
    }

    // 4) projection GEMM (A = fp32 accumulator converted inline) + bias, f32 out
    wmma_gemm<true, true><<<dim3(M / BM, C / BN), 256, 0, stream>>>(
        nullptr, accp, wprojT, nullptr, out, bproj, C, C);
}
